// GaussianHyperGru_33921651704152
// MI455X (gfx1250) — matrix operand order repaired and checked by reading, not compiled, not run on hardware
//
#include <hip/hip_runtime.h>
#include <hip/hip_bf16.h>

// ---------------- problem constants ----------------
constexpr int kIN = 64, kOD = 8, kPI = 48;
constexpr int kH = 256, kL = 32, kP = 16;
constexpr int kB = 128, kT = 512;
constexpr int kG3 = 3 * kH;          // 768 gate columns
constexpr int kNPAR = 1056;          // 768 + 16 + 256 + 16
constexpr int kGHLD = 784;           // 768 + 16 pad (LDS bank spread)

// d_out layout: mean (B*T*8) | std (B*T*8) | hT (B*256)
constexpr size_t kMeanElems = (size_t)kB * kT * kOD;   // 524288
constexpr size_t kStdOff    = kMeanElems;
constexpr size_t kHTOff     = 2 * kMeanElems;          // 1048576

// workspace layout (bytes)
constexpr size_t kTeBytes  = (size_t)kB * kT * kL * 4; // 8,388,608 task_enc f32
constexpr size_t kWihOff   = kTeBytes;                 // 768*64 bf16
constexpr size_t kWhhOff   = kWihOff  + (size_t)kG3 * kIN * 2;
constexpr size_t kWlatOff  = kWhhOff  + (size_t)kG3 * kH  * 2;
constexpr size_t kWparOff  = kWlatOff + (size_t)kL  * kH  * 2;

typedef __bf16 bf16_t;
typedef __bf16 v16bf __attribute__((ext_vector_type(16)));
typedef float  v8f   __attribute__((ext_vector_type(8)));

struct U128 { unsigned int w[4]; };
struct U256 { U128 lo, hi; };

static __device__ inline v16bf pack32B(U128 lo, U128 hi) {
    U256 t{lo, hi};
    return __builtin_bit_cast(v16bf, t);
}

// WMMA D = A(16x32 bf16) * B(32x16 bf16) + C(16x16 f32)
static __device__ inline v8f wmma_bf16(v16bf a, v16bf b, v8f c) {
    return __builtin_amdgcn_wmma_f32_16x16x32_bf16(false, a, false, b,
                                                   (short)0, c, false, false);
}

// B fragment: B[k][n] = W[n0+n][k0+k], W row-major bf16 with leading dim ldk.
static __device__ inline v16bf load_b_frag(const bf16_t* W, int ldk, int n0,
                                           int k0, int lane) {
    const bf16_t* p = W + (size_t)(n0 + (lane & 15)) * ldk + k0
                        + ((lane & 16) ? 8 : 0);
    return pack32B(*(const U128*)p, *(const U128*)(p + 16));
}

// A fragment from LDS bf16 tile (row-major, leading dim ld), tile K-base k0.
static __device__ inline v16bf load_a_frag_lds(const bf16_t* base, int ld,
                                               int k0, int lane) {
    const bf16_t* p = base + (lane & 15) * ld + k0 + ((lane & 16) ? 8 : 0);
    return pack32B(*(const U128*)p, *(const U128*)(p + 16));
}

// A fragment from an f32 row (per-lane row pointer), converting to bf16.
static __device__ inline v16bf a_frag_from_f32row(const float* rowp, int k0,
                                                  int lane) {
    int koff = (lane & 16) ? 8 : 0;
    v16bf a;
#pragma unroll
    for (int e = 0; e < 8; ++e) {
        a[e]     = (bf16_t)rowp[k0 + koff + e];
        a[e + 8] = (bf16_t)rowp[k0 + 16 + koff + e];
    }
    return a;
}

static __device__ inline float sigmoidf(float v) {
    return 1.0f / (1.0f + expf(-v));
}

// ---------------- weight conversion f32 -> bf16 ----------------
__global__ void __launch_bounds__(256) prep_weights_kernel(
    const float* w_ih, const float* w_hh, const float* w_lat, const float* w_par,
    bf16_t* o_ih, bf16_t* o_hh, bf16_t* o_lat, bf16_t* o_par) {
    int i = blockIdx.x * 256 + threadIdx.x;
    if (i < kG3 * kIN) o_ih[i]  = (bf16_t)w_ih[i];
    if (i < kG3 * kH)  o_hh[i]  = (bf16_t)w_hh[i];
    if (i < kL * kH)   o_lat[i] = (bf16_t)w_lat[i];
    if (i < kNPAR * kL) o_par[i] = (bf16_t)w_par[i];
}

// ---------------- fused GRU scan + latent projection (persistent weights) ----
// grid = B/16 workgroups of 256 threads (8 waves -> 512-VGPR budget/wave).
// Each wave owns 6 of the 48 gate N-tiles; w_hh fragments for 5 tiles stay
// register-resident all scan (320 VGPRs); the 6th tile + w_ih + w_lat are
// streamed from L2 each step through an opaque loop-variant pointer so LICM
// cannot hoist them into spills.
__global__ void __launch_bounds__(256, 1) gru_scan_kernel(
    const float* __restrict__ x, const float* __restrict__ hidden0,
    const float* __restrict__ b_ih, const float* __restrict__ b_hh,
    const float* __restrict__ b_lat,
    const bf16_t* __restrict__ w_ih_bf, const bf16_t* __restrict__ w_hh_bf,
    const bf16_t* __restrict__ w_lat_bf,
    float* __restrict__ task_enc, float* __restrict__ hT_out) {
    extern __shared__ char smem[];
    float*  gh  = (float*)smem;                      // [16][kGHLD] h-part
    float*  gx  = gh + 16 * kGHLD;                   // [16][kGHLD] x-part
    float*  hf  = gx + 16 * kGHLD;                   // [16][256] h (f32)
    float*  biL = hf + 16 * kH;                      // [768]
    float*  bhL = biL + kG3;                         // [768]
    float*  blL = bhL + kG3;                         // [32]
    bf16_t* hb  = (bf16_t*)(blL + kL);               // [16][256] h (bf16)

    const int tid  = threadIdx.x;
    const int lane = tid & 31;
    const int wave = tid >> 5;                       // 0..7
    const int r0   = blockIdx.x * 16;                // batch rows [r0, r0+16)

    for (int i = tid; i < kG3; i += 256) { biL[i] = b_ih[i]; bhL[i] = b_hh[i]; }
    if (tid < kL) blL[tid] = b_lat[tid];
    for (int i = tid; i < 16 * kH; i += 256) {
        int m = i >> 8, j = i & 255;
        float v = hidden0[(size_t)(r0 + m) * kH + j];
        hf[m * kH + j] = v;
        hb[m * kH + j] = (bf16_t)v;
    }

    // Register-resident w_hh fragments for tiles 0..4 of this wave (320 VGPRs).
    v16bf whh[5][8];
#pragma unroll
    for (int i = 0; i < 5; ++i) {
        int n0 = (wave * 6 + i) * 16;
#pragma unroll
        for (int kc = 0; kc < 8; ++kc)
            whh[i][kc] = load_b_frag(w_hh_bf, kH, n0, kc * 32, lane);
    }
    __syncthreads();

#pragma unroll 1
    for (int t = 0; t < kT; ++t) {
        // Opaque zero re-materialized every iteration: derived pointers are
        // loop-variant, so streamed weight loads cannot be hoisted out of t.
        int zoff = 0;
        asm volatile("" : "+v"(zoff));
        const bf16_t* whh_s = w_hh_bf + zoff;
        const bf16_t* wih_s = w_ih_bf + zoff;
        const bf16_t* wlat_s = w_lat_bf + zoff;

        // A fragments for h (K = 256 -> 8 chunks) from LDS
        v16bf ah[8];
#pragma unroll
        for (int kc = 0; kc < 8; ++kc)
            ah[kc] = load_a_frag_lds(hb, kH, kc * 32, lane);
        // A fragments for x_t (K = 64 -> 2 chunks), converted f32->bf16
        const float* xrow = x + ((size_t)(r0 + (lane & 15)) * kT + t) * kIN;
        v16bf ax0 = a_frag_from_f32row(xrow, 0, lane);
        v16bf ax1 = a_frag_from_f32row(xrow, 32, lane);

#pragma unroll
        for (int i = 0; i < 6; ++i) {
            int n0 = (wave * 6 + i) * 16;
            v8f ch = {};
            v8f cx = {};
#pragma unroll
            for (int kc = 0; kc < 8; ++kc) {
                v16bf b = (i < 5) ? whh[i][kc]
                                  : load_b_frag(whh_s, kH, n0, kc * 32, lane);
                ch = wmma_bf16(ah[kc], b, ch);
            }
            cx = wmma_bf16(ax0, load_b_frag(wih_s, kIN, n0, 0, lane), cx);
            cx = wmma_bf16(ax1, load_b_frag(wih_s, kIN, n0, 32, lane), cx);

            int col = n0 + (lane & 15);
            int mhi = (lane & 16) ? 8 : 0;
#pragma unroll
            for (int r = 0; r < 8; ++r) {
                gh[(r + mhi) * kGHLD + col] = ch[r];
                gx[(r + mhi) * kGHLD + col] = cx[r];
            }
        }
        __syncthreads();

        // gate nonlinearity: 16 rows x 256 cols over 256 threads (16 each)
        {
            int m  = tid >> 4;       // row 0..15
            int jb = tid & 15;
#pragma unroll 4
            for (int q = 0; q < 16; ++q) {
                int j = jb * 16 + q;
                float pr = gx[m * kGHLD + j] + biL[j]
                         + gh[m * kGHLD + j] + bhL[j];
                float pz = gx[m * kGHLD + j + kH] + biL[j + kH]
                         + gh[m * kGHLD + j + kH] + bhL[j + kH];
                float xn = gx[m * kGHLD + j + 2 * kH] + biL[j + 2 * kH];
                float hn = gh[m * kGHLD + j + 2 * kH] + bhL[j + 2 * kH];
                float rg = sigmoidf(pr);
                float zg = sigmoidf(pz);
                float ng = tanhf(xn + rg * hn);
                float hv = (1.0f - zg) * ng + zg * hf[m * kH + j];
                hf[m * kH + j] = hv;
                hb[m * kH + j] = (bf16_t)hv;
            }
        }
        __syncthreads();

        // task_enc tile (16x32) = h_new @ w_lat^T  (waves 0,1; overlaps next
        // step's WMMA phase — hb is read-only until the next gate barrier)
        if (wave < 2) {
            int n0 = wave * 16;
            v8f c = {};
#pragma unroll
            for (int kc = 0; kc < 8; ++kc) {
                v16bf a = load_a_frag_lds(hb, kH, kc * 32, lane);
                v16bf b = load_b_frag(wlat_s, kH, n0, kc * 32, lane);
                c = wmma_bf16(a, b, c);
            }
            int col = n0 + (lane & 15);
            int mhi = (lane & 16) ? 8 : 0;
            float blv = blL[col];
#pragma unroll
            for (int r = 0; r < 8; ++r) {
                int m = r + mhi;
                task_enc[((size_t)(r0 + m) * kT + t) * kL + col] = c[r] + blv;
            }
        }
    }
    __syncthreads();
    for (int i = tid; i < 16 * kH; i += 256) {
        int m = i >> 8, j = i & 255;
        hT_out[(size_t)(r0 + m) * kH + j] = hf[m * kH + j];
    }
}

// ---------------- fused hypernet: params GEMM + per-row MLP ----------------
__global__ void __launch_bounds__(256) hyper_kernel(
    const float* __restrict__ x, const float* __restrict__ task_enc,
    const bf16_t* __restrict__ w_par_bf, const float* __restrict__ b_par,
    float* __restrict__ out_mean, float* __restrict__ out_std) {
    extern __shared__ char smem[];
    float* par = (float*)smem;                 // [16][1056]
    float* oh  = par + 16 * kNPAR;             // [16][16]

    const int tid  = threadIdx.x;
    const int lane = tid & 31;
    const int wave = tid >> 5;
    const size_t row0 = (size_t)blockIdx.x * 16;   // flattened b*T+t rows

    // A fragment: 16 task_enc rows, K=32 exactly one WMMA chunk
    const float* terow = task_enc + (row0 + (lane & 15)) * kL;
    v16bf a = a_frag_from_f32row(terow, 0, lane);

    for (int nt = wave; nt < kNPAR / 16; nt += 8) {   // 66 N-tiles
        int n0 = nt * 16;
        v8f c = {};
        c = wmma_bf16(a, load_b_frag(w_par_bf, kL, n0, 0, lane), c);
        int col = n0 + (lane & 15);
        int mhi = (lane & 16) ? 8 : 0;
        float bp = b_par[col];
#pragma unroll
        for (int r = 0; r < 8; ++r)
            par[(r + mhi) * kNPAR + col] = c[r] + bp;
    }
    __syncthreads();

    const int m = tid >> 4;
    const size_t rowg = row0 + m;
    const float* xr = x + rowg * kIN;          // states = x[..., :48]
    {
        int p = tid & 15;
        float acc = par[m * kNPAR + 768 + p];  // b_h
#pragma unroll
        for (int i = 0; i < kPI; ++i)
            acc += par[m * kNPAR + p * kPI + i] * xr[i];
        oh[m * 16 + p] = tanhf(acc);
    }
    __syncthreads();
    {
        int o = tid & 15;                      // 2*OD = 16 outputs
        float acc = par[m * kNPAR + 1040 + o]; // b_o
#pragma unroll
        for (int p = 0; p < kP; ++p)
            acc += par[m * kNPAR + 784 + o * 16 + p] * oh[m * 16 + p];
        if (o < kOD) out_mean[rowg * kOD + o] = acc;
        else         out_std[rowg * kOD + (o - kOD)] = expf(acc);
    }
}

extern "C" void kernel_launch(void* const* d_in, const int* in_sizes, int n_in,
                              void* d_out, int out_size, void* d_ws, size_t ws_size,
                              hipStream_t stream) {
    const float* x      = (const float*)d_in[0];
    const float* hidden = (const float*)d_in[1];
    const float* w_ih   = (const float*)d_in[2];
    const float* w_hh   = (const float*)d_in[3];
    const float* b_ih   = (const float*)d_in[4];
    const float* b_hh   = (const float*)d_in[5];
    const float* w_lat  = (const float*)d_in[6];
    const float* b_lat  = (const float*)d_in[7];
    const float* w_par  = (const float*)d_in[8];
    const float* b_par  = (const float*)d_in[9];
    float* out = (float*)d_out;
    char*  ws  = (char*)d_ws;

    float*  te      = (float*)ws;
    bf16_t* wih_bf  = (bf16_t*)(ws + kWihOff);
    bf16_t* whh_bf  = (bf16_t*)(ws + kWhhOff);
    bf16_t* wlat_bf = (bf16_t*)(ws + kWlatOff);
    bf16_t* wpar_bf = (bf16_t*)(ws + kWparOff);

    // 1) weights -> bf16 (largest array = 196608 elements)
    prep_weights_kernel<<<dim3((kG3 * kH + 255) / 256), dim3(256), 0, stream>>>(
        w_ih, w_hh, w_lat, w_par, wih_bf, whh_bf, wlat_bf, wpar_bf);

    // 2) persistent-weight fused GRU scan (+gx on the fly, +latent proj), 8 WGs
    size_t scan_smem =
        (size_t)(2 * 16 * kGHLD + 16 * kH + 2 * kG3 + kL) * 4   // f32 arrays
        + (size_t)16 * kH * 2;                                  // hb (bf16)
    gru_scan_kernel<<<dim3(kB / 16), dim3(256), scan_smem, stream>>>(
        x, hidden, b_ih, b_hh, b_lat, wih_bf, whh_bf, wlat_bf,
        te, out + kHTOff);

    // 3) fused hypernet params GEMM + MLP head, 4096 WGs
    size_t hyper_smem = (size_t)(16 * kNPAR + 16 * 16) * 4;
    hyper_kernel<<<dim3((kB * kT) / 16), dim3(256), hyper_smem, stream>>>(
        x, te, wpar_bf, b_par, out, out + kStdOff);
}
